// GNNBlock_69664369541133
// MI455X (gfx1250) — matrix operand order repaired and checked by reading, not compiled, not run on hardware
//
#include <hip/hip_runtime.h>
#include <hip/hip_bf16.h>
#include <stdint.h>

typedef __attribute__((ext_vector_type(16))) __bf16 v16bf;
typedef __attribute__((ext_vector_type(2)))  __bf16 v2bf;
typedef __attribute__((ext_vector_type(8)))  float  v8f;

union BFrag { v16bf v; uint4 q[2]; unsigned int w[8]; unsigned short u[16]; };
union FAcc  { v8f v; float f[8]; };

#define BB 8
#define NT 1024
#define NS 1024
#define DD 128
#define HH 4

// pack two f32 -> two bf16 (RNE via hardware fptrunc)
__device__ __forceinline__ unsigned int f2bf_pk(float lo, float hi) {
  v2bf r;
  r.x = (__bf16)lo;
  r.y = (__bf16)hi;
  return __builtin_bit_cast(unsigned int, r);
}

__device__ __forceinline__ unsigned short f2bf1(float f) {
  return __builtin_bit_cast(unsigned short, (__bf16)f);
}

__device__ __forceinline__ v8f wmma_bf16(v16bf a, v16bf b, v8f c) {
  return __builtin_amdgcn_wmma_f32_16x16x32_bf16(false, a, false, b, (short)0, c,
                                                 false, false);
}

// ---------------------------------------------------------------------------
// K0: fold attention vectors into projection weights:
//   wt[h*128+k] = sum_d W[h*128+d, k] * att[h*128+d]
// ---------------------------------------------------------------------------
__global__ void k_wtilde(const float* __restrict__ Wsrc, const float* __restrict__ Wtgt,
                         const float* __restrict__ attS, const float* __restrict__ attT,
                         float* __restrict__ wtS, float* __restrict__ wtT) {
  int tid = threadIdx.x;            // 0..511
  int h = tid >> 7, k = tid & 127;
  float a1 = 0.f, a2 = 0.f;
  for (int d = 0; d < 128; ++d) {
    int o = h * 128 + d;
    a1 += Wsrc[(size_t)o * 128 + k] * attS[o];
    a2 += Wtgt[(size_t)o * 128 + k] * attT[o];
  }
  wtS[tid] = a1;
  wtT[tid] = a2;
}

// ---------------------------------------------------------------------------
// K0b: one-time f32 -> bf16 conversion of the three weight matrices.
//   [0, 65536)           W_src      -> wsrcB
//   [65536, 98304)       W_gate_val -> wgvB
//   [98304, 114688)      W_out      -> woutB
// ---------------------------------------------------------------------------
__global__ void k_cvtw(const float* __restrict__ Wsrc, const float* __restrict__ Wgv,
                       const float* __restrict__ Wout,
                       unsigned short* __restrict__ wsrcB,
                       unsigned short* __restrict__ wgvB,
                       unsigned short* __restrict__ woutB) {
  int idx = blockIdx.x * 256 + threadIdx.x;    // < 114688
  if (idx < 65536) {
    wsrcB[idx] = f2bf1(Wsrc[idx]);
  } else if (idx < 98304) {
    int j = idx - 65536;
    wgvB[j] = f2bf1(Wgv[j]);
  } else {
    int j = idx - 98304;
    woutB[j] = f2bf1(Wout[j]);
  }
}

// ---------------------------------------------------------------------------
// K1: a[b,h,n] = x[b,n,:] . wt[h,:]      (layout [B,H,N])
// ---------------------------------------------------------------------------
__global__ void k_avec(const float* __restrict__ x, const float* __restrict__ wt,
                       float* __restrict__ aOut) {
  int idx = blockIdx.x * 256 + threadIdx.x;   // = b*4096 + h*1024 + n
  int n = idx & 1023, h = (idx >> 10) & 3, b = idx >> 12;
  const float* row = x + (size_t)(b * 1024 + n) * 128;
  const float* w = wt + h * 128;
  float acc = 0.f;
  for (int k = 0; k < 128; ++k) acc += row[k] * w[k];
  aOut[idx] = acc;
}

// ---------------------------------------------------------------------------
// K2: h_src = x_source @ W_src^T  via bf16 WMMA, stored bf16 TRANSPOSED as
//     hT[b][h][d][s]  so aggregation B-fragments are contiguous in s.
// grid = B * (NS/16) blocks of 256 threads; wave w covers o-tiles {w,w+8,w+16,w+24}
// ---------------------------------------------------------------------------
__global__ __launch_bounds__(256) void k_hsrc(const float* __restrict__ xs,
                                              const unsigned short* __restrict__ wsrcB,
                                              unsigned short* __restrict__ hT) {
  int b = blockIdx.x >> 6;
  int sbase = (blockIdx.x & 63) * 16;
  int w = threadIdx.x >> 5, lane = threadIdx.x & 31;
  int m = lane & 15, hi = lane >> 4;

  // A fragments (16x32 bf16) for all 4 K-chunks, reused across o-tiles
  BFrag af[4];
  const float* xrow = xs + (size_t)(b * NS + sbase + m) * DD;
  for (int kk = 0; kk < 4; ++kk) {
    int k0 = kk * 32 + hi * 8;
#pragma unroll
    for (int e = 0; e < 4; ++e)
      af[kk].w[e] = f2bf_pk(xrow[k0 + 2 * e], xrow[k0 + 2 * e + 1]);
#pragma unroll
    for (int e = 0; e < 4; ++e)
      af[kk].w[4 + e] = f2bf_pk(xrow[k0 + 16 + 2 * e], xrow[k0 + 16 + 2 * e + 1]);
  }

  for (int rep = 0; rep < 4; ++rep) {
    int ot = w + rep * 8;          // 0..31 output col tiles (512 cols)
    int o = ot * 16 + m;           // this lane's output column (B-frag col)
    FAcc acc;
#pragma unroll
    for (int i = 0; i < 8; ++i) acc.f[i] = 0.f;
    for (int kk = 0; kk < 4; ++kk) {
      BFrag bf;  // B (32x16): lane = col, elements = k at hi*8..+7 and 16+hi*8..+7
      const unsigned short* wrow = wsrcB + (size_t)o * DD + kk * 32 + hi * 8;
      bf.q[0] = *(const uint4*)wrow;
      bf.q[1] = *(const uint4*)(wrow + 16);
      acc.v = wmma_bf16(af[kk].v, bf.v, acc.v);
    }
    // D layout: row = r + 8*hi, col = m. Transposed store -> contiguous in s.
    int h = o >> 7, d = o & 127;
    union { uint4 q; unsigned int w[4]; } st;
#pragma unroll
    for (int r = 0; r < 4; ++r) st.w[r] = f2bf_pk(acc.f[2 * r], acc.f[2 * r + 1]);
    *(uint4*)(hT + (size_t)((b * HH + h) * DD + d) * NS + sbase + hi * 8) = st.q;
  }
}

// ---------------------------------------------------------------------------
// K3: masked online softmax stats per (b,t,h): row max + 1/denominator.
// One wave per (b,t); 4 heads in registers.
// ---------------------------------------------------------------------------
__global__ __launch_bounds__(256) void k_rowstats(
    const float* __restrict__ adj, const float* __restrict__ aS,
    const float* __restrict__ aT, float* __restrict__ rmx, float* __restrict__ rdn) {
  int gw = blockIdx.x * 8 + (threadIdx.x >> 5);
  int lane = threadIdx.x & 31;
  int b = gw >> 10, t = gw & 1023;
  float at[4], mx[4], sm[4];
#pragma unroll
  for (int h = 0; h < 4; ++h) {
    at[h] = aT[(size_t)(b * HH + h) * NT + t];
    mx[h] = -1e30f;
    sm[h] = 0.f;
  }
  const float* arow = adj + (size_t)(b * NT + t) * NS;
  for (int s = lane; s < NS; s += 32) {
    float adjv = arow[s];
    if (adjv != 0.f) {
#pragma unroll
      for (int h = 0; h < 4; ++h) {
        float x = at[h] + aS[(size_t)(b * HH + h) * NS + s];
        x = x >= 0.f ? x : 0.2f * x;   // leaky relu
        float nm = fmaxf(mx[h], x);
        sm[h] = sm[h] * __expf(mx[h] - nm) + __expf(x - nm);
        mx[h] = nm;
      }
    }
  }
#pragma unroll
  for (int off = 1; off < 32; off <<= 1) {
#pragma unroll
    for (int h = 0; h < 4; ++h) {
      float om = __shfl_xor(mx[h], off, 32);
      float os = __shfl_xor(sm[h], off, 32);
      float nm = fmaxf(mx[h], om);
      sm[h] = sm[h] * __expf(mx[h] - nm) + os * __expf(om - nm);
      mx[h] = nm;
    }
  }
  if (lane == 0) {
#pragma unroll
    for (int h = 0; h < 4; ++h) {
      size_t ix = (size_t)(b * HH + h) * NT + t;
      rmx[ix] = mx[h];
      rdn[ix] = sm[h] > 0.f ? 1.f / sm[h] : 0.f;
    }
  }
}

// ---------------------------------------------------------------------------
// K4: fused attention aggregation + head-mean + bias + residual + LN1 +
//     SwiGLU FFN + residual + LN2.   One block per (b, 16-row t-tile).
// 8 waves; wave w owns output d-columns [16w, 16w+16).
// ---------------------------------------------------------------------------
__global__ __launch_bounds__(256) void k_fused(
    const float* __restrict__ xt, const float* __restrict__ adj,
    const unsigned short* __restrict__ hT,
    const float* __restrict__ aS, const float* __restrict__ aTt,
    const float* __restrict__ rmx, const float* __restrict__ rdn,
    const float* __restrict__ gbias,
    const float* __restrict__ ln1g, const float* __restrict__ ln1b,
    const unsigned short* __restrict__ wgvB,
    const unsigned short* __restrict__ woutB,
    const float* __restrict__ ln2g, const float* __restrict__ ln2b,
    float* __restrict__ out) {
  const int XP = 132;   // padded row stride (floats) to spread LDS banks
  const int GVP = 264;
  __shared__ __align__(16) unsigned short attnb[2][16 * 32];
  __shared__ float s_at[4][16], s_mx[4][16], s_rd[4][16];
  __shared__ float xf32[16 * 132];
  __shared__ __align__(16) unsigned short xbf[16 * 128];
  __shared__ float gvL[16 * 264];
  __shared__ __align__(16) unsigned short hbf[16 * 128];

  int b = blockIdx.x >> 6;
  int tbase = (blockIdx.x & 63) * 16;
  int tid = threadIdx.x;
  int w = tid >> 5, lane = tid & 31;
  int n = lane & 15, hi = lane >> 4;
  int dbase = w * 16;

  if (tid < 64) {
    int h = tid >> 4, t = tid & 15;
    size_t ix = (size_t)(b * HH + h) * NT + tbase + t;
    s_at[h][t] = aTt[ix];
    s_mx[h][t] = rmx[ix];
    s_rd[h][t] = rdn[ix];
  }
  __syncthreads();

  // ---- attention aggregation: acc += sum_h attn_h @ hsrc_h ----
  FAcc acc;
#pragma unroll
  for (int i = 0; i < 8; ++i) acc.f[i] = 0.f;

  // thread -> (row, 2 adjacent cols) map for the 16x32 attn tile
  int a_t = tid >> 4;            // 0..15
  int a_c = (tid & 15) * 2;      // 0,2,..,30

  for (int h = 0; h < 4; ++h) {
    float s_atv = s_at[h][a_t];
    float s_mxv = s_mx[h][a_t];
    float s_rdv = s_rd[h][a_t];
    for (int sc = 0; sc < 32; ++sc) {
      int buf = (h * 32 + sc) & 1;
      // cooperatively build 16x32 bf16 attention tile (2 adjacent cols/thread)
      {
        int s0 = sc * 32 + a_c;
        float2 adj2 = *(const float2*)&adj[(size_t)(b * NT + tbase + a_t) * NS + s0];
        float2 as2  = *(const float2*)&aS[(size_t)(b * HH + h) * NS + s0];
        float w0 = 0.f, w1 = 0.f;
        if (adj2.x != 0.f) {
          float x = s_atv + as2.x;
          x = x >= 0.f ? x : 0.2f * x;
          w0 = __expf(x - s_mxv) * s_rdv;
        }
        if (adj2.y != 0.f) {
          float x = s_atv + as2.y;
          x = x >= 0.f ? x : 0.2f * x;
          w1 = __expf(x - s_mxv) * s_rdv;
        }
        *(unsigned int*)&attnb[buf][a_t * 32 + a_c] = f2bf_pk(w0, w1);
      }
      __syncthreads();
      BFrag A, Bf;
      A.q[0] = *(const uint4*)&attnb[buf][n * 32 + hi * 8];
      A.q[1] = *(const uint4*)&attnb[buf][n * 32 + 16 + hi * 8];
      const unsigned short* bs =
          hT + (size_t)((b * HH + h) * DD + dbase + n) * NS + sc * 32 + hi * 8;
      Bf.q[0] = *(const uint4*)bs;
      Bf.q[1] = *(const uint4*)(bs + 16);
      acc.v = wmma_bf16(A.v, Bf.v, acc.v);
    }
  }

  // ---- head mean + bias + residual -> xf32 ----
  {
    int d = dbase + n;
    float bias = gbias[d];
#pragma unroll
    for (int r = 0; r < 8; ++r) {
      int t = r + hi * 8;
      float p = acc.f[r] * 0.25f + bias +
                xt[(size_t)(b * NT + tbase + t) * DD + d];
      xf32[t * XP + d] = p;
    }
  }
  __syncthreads();

  // ---- LayerNorm 1 (in place) + bf16 copy for FFN ----
  {
    int t = tid >> 4, c0 = (tid & 15) * 8;   // 8 adjacent cols per thread
    float sm = 0.f, sq = 0.f;
#pragma unroll
    for (int i = 0; i < 8; ++i) {
      float v = xf32[t * XP + c0 + i];
      sm += v; sq += v * v;
    }
#pragma unroll
    for (int off = 1; off < 16; off <<= 1) {
      sm += __shfl_xor(sm, off, 32);
      sq += __shfl_xor(sq, off, 32);
    }
    float mean = sm * (1.f / 128.f);
    float var = sq * (1.f / 128.f) - mean * mean;
    float rstd = rsqrtf(var + 1e-5f);
    float vv[8];
#pragma unroll
    for (int i = 0; i < 8; ++i) {
      int c = c0 + i;
      vv[i] = (xf32[t * XP + c] - mean) * rstd * ln1g[c] + ln1b[c];
      xf32[t * XP + c] = vv[i];
    }
#pragma unroll
    for (int i = 0; i < 4; ++i)
      *(unsigned int*)&xbf[t * 128 + c0 + 2 * i] = f2bf_pk(vv[2 * i], vv[2 * i + 1]);
  }
  __syncthreads();

  // ---- FFN GEMM1: gv = x @ Wgv^T (N=256: wave w does tiles w and w+8) ----
  for (int rep = 0; rep < 2; ++rep) {
    int ot = w + rep * 8;
    FAcc g1;
#pragma unroll
    for (int i = 0; i < 8; ++i) g1.f[i] = 0.f;
    for (int kk = 0; kk < 4; ++kk) {
      BFrag A, Bf;
      A.q[0] = *(const uint4*)&xbf[n * 128 + kk * 32 + hi * 8];
      A.q[1] = *(const uint4*)&xbf[n * 128 + kk * 32 + 16 + hi * 8];
      const unsigned short* wrow =
          wgvB + (size_t)(ot * 16 + n) * 128 + kk * 32 + hi * 8;
      Bf.q[0] = *(const uint4*)wrow;
      Bf.q[1] = *(const uint4*)(wrow + 16);
      g1.v = wmma_bf16(A.v, Bf.v, g1.v);
    }
#pragma unroll
    for (int r = 0; r < 8; ++r)
      gvL[(r + hi * 8) * GVP + ot * 16 + n] = g1.f[r];
  }
  __syncthreads();

  // ---- hidden = val * silu(gate) ----
  {
    int t = tid >> 4, j0 = (tid & 15) * 8;
    float hz[8];
#pragma unroll
    for (int i = 0; i < 8; ++i) {
      int j = j0 + i;
      float v = gvL[t * GVP + j];
      float g = gvL[t * GVP + 128 + j];
      hz[i] = v * (g / (1.f + __expf(-g)));
    }
#pragma unroll
    for (int i = 0; i < 4; ++i)
      *(unsigned int*)&hbf[t * 128 + j0 + 2 * i] = f2bf_pk(hz[2 * i], hz[2 * i + 1]);
  }
  __syncthreads();

  // ---- FFN GEMM2: hidden @ Wout^T, residual into xf32 ----
  {
    FAcc g2;
#pragma unroll
    for (int i = 0; i < 8; ++i) g2.f[i] = 0.f;
    for (int kk = 0; kk < 4; ++kk) {
      BFrag A, Bf;
      A.q[0] = *(const uint4*)&hbf[n * 128 + kk * 32 + hi * 8];
      A.q[1] = *(const uint4*)&hbf[n * 128 + kk * 32 + 16 + hi * 8];
      const unsigned short* wrow =
          woutB + (size_t)(dbase + n) * 128 + kk * 32 + hi * 8;
      Bf.q[0] = *(const uint4*)wrow;
      Bf.q[1] = *(const uint4*)(wrow + 16);
      g2.v = wmma_bf16(A.v, Bf.v, g2.v);
    }
#pragma unroll
    for (int r = 0; r < 8; ++r) {
      int t = r + hi * 8;
      xf32[t * XP + dbase + n] += g2.f[r];
    }
  }
  __syncthreads();

  // ---- LayerNorm 2 -> output ----
  {
    int t = tid >> 4, c0 = (tid & 15) * 8;
    float sm = 0.f, sq = 0.f;
#pragma unroll
    for (int i = 0; i < 8; ++i) {
      float v = xf32[t * XP + c0 + i];
      sm += v; sq += v * v;
    }
#pragma unroll
    for (int off = 1; off < 16; off <<= 1) {
      sm += __shfl_xor(sm, off, 32);
      sq += __shfl_xor(sq, off, 32);
    }
    float mean = sm * (1.f / 128.f);
    float var = sq * (1.f / 128.f) - mean * mean;
    float rstd = rsqrtf(var + 1e-5f);
#pragma unroll
    for (int i = 0; i < 8; ++i) {
      int c = c0 + i;
      float v = (xf32[t * XP + c] - mean) * rstd * ln2g[c] + ln2b[c];
      out[(size_t)(b * NT + tbase + t) * DD + c] = v;
    }
  }
}

// ---------------------------------------------------------------------------
extern "C" void kernel_launch(void* const* d_in, const int* in_sizes, int n_in,
                              void* d_out, int out_size, void* d_ws, size_t ws_size,
                              hipStream_t stream) {
  (void)in_sizes; (void)n_in; (void)out_size; (void)ws_size;
  const float* x_tgt = (const float*)d_in[0];
  const float* adj   = (const float*)d_in[1];
  const float* x_src = (const float*)d_in[2];
  const float* Wsrc  = (const float*)d_in[3];
  const float* Wtgt  = (const float*)d_in[4];
  const float* attS  = (const float*)d_in[5];
  const float* attT  = (const float*)d_in[6];
  const float* gbias = (const float*)d_in[7];
  const float* ln1g  = (const float*)d_in[8];
  const float* ln1b  = (const float*)d_in[9];
  const float* Wgv   = (const float*)d_in[10];
  const float* Wout  = (const float*)d_in[11];
  const float* ln2g  = (const float*)d_in[12];
  const float* ln2b  = (const float*)d_in[13];
  float* out = (float*)d_out;

  // workspace layout: f32 region, then bf16 regions (~9.1 MB total)
  float* ws = (float*)d_ws;
  float* wtS = ws;                       // 512
  float* wtT = ws + 512;                 // 512
  float* aS  = ws + 1024;                // B*H*NS = 32768
  float* aTt = ws + 1024 + 32768;        // 32768
  float* rmx = ws + 1024 + 65536;        // 32768
  float* rdn = ws + 1024 + 98304;        // 32768
  unsigned short* hT    = (unsigned short*)(ws + 132096);  // B*H*D*NS = 4194304
  unsigned short* wsrcB = hT + 4194304;                    // 65536
  unsigned short* wgvB  = wsrcB + 65536;                   // 32768
  unsigned short* woutB = wgvB + 32768;                    // 16384

  k_wtilde<<<1, 512, 0, stream>>>(Wsrc, Wtgt, attS, attT, wtS, wtT);
  k_cvtw<<<448, 256, 0, stream>>>(Wsrc, Wgv, Wout, wsrcB, wgvB, woutB);
  k_avec<<<128, 256, 0, stream>>>(x_src, wtS, aS);
  k_avec<<<128, 256, 0, stream>>>(x_tgt, wtT, aTt);
  k_hsrc<<<BB * (NS / 16), 256, 0, stream>>>(x_src, wsrcB, hT);
  k_rowstats<<<(BB * NT) / 8, 256, 0, stream>>>(adj, aS, aTt, rmx, rdn);
  k_fused<<<BB * (NT / 16), 256, 0, stream>>>(x_tgt, adj, hT, aS, aTt, rmx, rdn,
                                              gbias, ln1g, ln1b, wgvB, woutB,
                                              ln2g, ln2b, out);
}
